// LlamaAttentionBiased_50173807952542
// MI455X (gfx1250) — matrix operand order repaired and checked by reading, not compiled, not run on hardware
//
#include <hip/hip_runtime.h>
#include <hip/hip_bf16.h>

// ---------------------------------------------------------------------------
// LlamaAttentionBiased on gfx1250 (MI455X): bf16 WMMA pipeline with
// async-to-LDS staged GEMMs and a wave32 flash-attention core.
//   B=1, S=2048, HID=2048, H=32, KVH=8, D=64, block-causal bias (bs=num_masks+1)
// ---------------------------------------------------------------------------

#define S_LEN 2048
#define HIDN  2048
#define NH    32
#define NKVH  8
#define HD    64

typedef __attribute__((ext_vector_type(16))) __bf16 v16bf;
typedef __attribute__((ext_vector_type(8)))  __bf16 v8bf;
typedef __attribute__((ext_vector_type(8)))  float  v8f;
typedef int v4i_vs __attribute__((vector_size(16)));   // matches async-LDS builtin pointee

// ---- CDNA5 async global->LDS path (guarded; falls back to sync copy) -------
#if defined(__gfx1250__) && __has_builtin(__builtin_amdgcn_global_load_async_to_lds_b128)
#define HAVE_ASYNC_LDS 1
#else
#define HAVE_ASYNC_LDS 0
#endif

#if HAVE_ASYNC_LDS
#if __has_builtin(__builtin_amdgcn_s_wait_asynccnt)
#define WAIT_ASYNC(n) __builtin_amdgcn_s_wait_asynccnt(n)
#else
#define WAIT_ASYNC(n) asm volatile("s_wait_asynccnt %0" ::"n"(n) : "memory")
#endif
#else
#define WAIT_ASYNC(n) ((void)0)
#endif

// ---- WMMA fragment loaders (CDNA5 wave32 layouts, 16-bit operands) ---------
// A-matrix 16x32 (MxK): lane%16 = M row.  lanes 0-15: e0..7 -> K 0..7,
// e8..15 -> K 16..23 ; lanes 16-31: +8 on each K chunk.
static __device__ __forceinline__
v16bf load_frag_a(const __bf16* base, int row, int ld, int col, int lane) {
  const __bf16* p = base + (size_t)(row + (lane & 15)) * ld + col + ((lane >> 4) << 3);
  union { v16bf v; v8bf h[2]; } u;
  u.h[0] = *(const v8bf*)(p);        // K = off+0..7
  u.h[1] = *(const v8bf*)(p + 16);   // K = off+16..23
  return u.v;
}
// B-matrix 32x16 (KxN): lane%16 = N col.  lanes 0-15: e0..7 -> K 0..7,
// e8..15 -> K 16..23 ; lanes 16-31: +8.  Source stored N-major (row = column of B).
static __device__ __forceinline__
v16bf load_frag_b(const __bf16* base, int row, int ld, int col, int lane) {
  const __bf16* p = base + (size_t)(row + (lane & 15)) * ld + col + ((lane >> 4) << 3);
  union { v16bf v; v8bf h[2]; } u;
  u.h[0] = *(const v8bf*)(p);
  u.h[1] = *(const v8bf*)(p + 16);
  return u.v;
}
// Same B layout, reading a staged tile in LDS (row pitch 40 halves = 80 B,
// keeps 16B alignment and spreads rows across banks).
#define BS_PITCH 40
static __device__ __forceinline__
v16bf lds_frag_b(const __bf16* Bsbuf, int trow, int lane) {
  const __bf16* p = Bsbuf + (trow + (lane & 15)) * BS_PITCH + ((lane >> 4) << 3);
  union { v16bf v; v8bf h[2]; } u;
  u.h[0] = *(const v8bf*)(p);
  u.h[1] = *(const v8bf*)(p + 16);
  return u.v;
}

#define WMMA_BF16(a, b, c) \
  __builtin_amdgcn_wmma_f32_16x16x32_bf16(false, (a), false, (b), (short)0, (c), false, false)

// ---- elementwise conversion -------------------------------------------------
__global__ void cvt_bf16_kernel(__bf16* __restrict__ dst, const float* __restrict__ src, int n) {
  int i = blockIdx.x * blockDim.x + threadIdx.x;
  if (i < n) dst[i] = (__bf16)src[i];
}

// W[K x N] (row-major) -> Wt[N x K] bf16, coalesced both ways via LDS tile.
// grid = (N/32, K/32), block = (32, 8)
__global__ void tcvt_bf16_kernel(__bf16* __restrict__ dst, const float* __restrict__ src,
                                 int N, int K) {
  __shared__ float tile[32][33];
  const int nb = blockIdx.x * 32, kb = blockIdx.y * 32;
  const int tx = threadIdx.x, ty = threadIdx.y;
#pragma unroll
  for (int j = 0; j < 32; j += 8)
    tile[ty + j][tx] = src[(size_t)(kb + ty + j) * N + nb + tx];
  __syncthreads();
#pragma unroll
  for (int j = 0; j < 32; j += 8)
    dst[(size_t)(nb + ty + j) * K + kb + tx] = (__bf16)tile[tx][ty + j];
}

// ---- WMMA GEMM: C[MxN] f32 = A[MxK] bf16 @ Bt[NxK]^T bf16 ------------------
// Block = 8 waves = 128 rows x 64 cols.  The 64x32 B tile is staged into LDS
// (double-buffered, async when available) and shared by all 8 waves; each
// wave streams its own A fragments straight from global (L2-resident).
__global__ void gemm_bt_kernel(float* __restrict__ C, const __bf16* __restrict__ A,
                               const __bf16* __restrict__ Bt, int M, int N, int K) {
  __shared__ __bf16 Bs[2][64 * BS_PITCH];

  const int lane = threadIdx.x;                       // 0..31
  const int wave = threadIdx.y;                       // 0..7
  const int m0 = (blockIdx.y * 8 + wave) * 16;
  const int n0 = blockIdx.x * 64;

  // staging decomposition: 256 threads x 16B = one 64x32 bf16 tile
  const int tid    = wave * 32 + lane;
  const int srow   = tid >> 2;                        // 0..63 (row of Bt tile)
  const int schunk = (tid & 3) * 8;                   // half-offset within row

  auto stage = [&](int buf, int k) {
    const __bf16* src = Bt + (size_t)(n0 + srow) * K + k + schunk;
    __bf16* dst = &Bs[buf][srow * BS_PITCH + schunk];
#if HAVE_ASYNC_LDS
    __builtin_amdgcn_global_load_async_to_lds_b128(
        (__attribute__((address_space(1))) v4i_vs*)src,
        (__attribute__((address_space(3))) v4i_vs*)dst, 0, 0);
#else
    *(v8bf*)dst = *(const v8bf*)src;
#endif
  };

  v8f acc[4] = {{}, {}, {}, {}};

  stage(0, 0);
  for (int k = 0, it = 0; k < K; k += 32, ++it) {
    const int cur = it & 1;
    const bool has_next = (k + 32) < K;
    if (has_next) stage(cur ^ 1, k + 32);
    if (has_next) { WAIT_ASYNC(1); } else { WAIT_ASYNC(0); }
    __syncthreads();

    v16bf a = load_frag_a(A, m0, K, k, lane);
    __builtin_prefetch(A + (size_t)(m0 + (lane & 15)) * K + k + 64, 0, 1);
#pragma unroll
    for (int t = 0; t < 4; ++t) {
      v16bf b = lds_frag_b(Bs[cur], t * 16, lane);
      acc[t] = WMMA_BF16(a, b, acc[t]);
    }
    __syncthreads();   // all waves done with buffer `cur` before it is re-staged
  }

  // C/D layout: col = n0 + t*16 + lane%16, rows = m0 + r + 8*(lane/16)
  const int cn = lane & 15;
  const int rb = m0 + ((lane >> 4) << 3);
#pragma unroll
  for (int t = 0; t < 4; ++t)
#pragma unroll
    for (int r = 0; r < 8; ++r)
      C[(size_t)(rb + r) * N + n0 + t * 16 + cn] = acc[t][r];
}

// ---- RoPE + repack ---------------------------------------------------------
// Qb[h][s][d] = rope(Q)[s,h,d] * D^-0.5   (bf16)
__global__ void rope_q_kernel(__bf16* __restrict__ Qb, const float* __restrict__ Qf,
                              const float* __restrict__ cosp, const float* __restrict__ sinp) {
  int s = blockIdx.x, h = blockIdx.y, d = threadIdx.x;
  size_t base = (size_t)s * (NH * HD) + h * HD;
  float x  = Qf[base + d];
  float xr = (d < 32) ? -Qf[base + d + 32] : Qf[base + d - 32];
  float c  = cosp[s * HD + d], sn = sinp[s * HD + d];
  Qb[(size_t)h * S_LEN * HD + (size_t)s * HD + d] = (__bf16)((x * c + xr * sn) * 0.125f);
}

// Kb[kv][s][d] = rope(K)[s,kv,d]   (bf16)
__global__ void rope_k_kernel(__bf16* __restrict__ Kb, const float* __restrict__ Kf,
                              const float* __restrict__ cosp, const float* __restrict__ sinp) {
  int s = blockIdx.x, kv = blockIdx.y, d = threadIdx.x;
  size_t base = (size_t)s * (NKVH * HD) + kv * HD;
  float x  = Kf[base + d];
  float xr = (d < 32) ? -Kf[base + d + 32] : Kf[base + d - 32];
  float c  = cosp[s * HD + d], sn = sinp[s * HD + d];
  Kb[(size_t)kv * S_LEN * HD + (size_t)s * HD + d] = (__bf16)(x * c + xr * sn);
}

// Vt[kv][d][s] = V[s,kv,d]   (bf16, d-major so PV B-operand loads are contiguous)
__global__ void vtrans_kernel(__bf16* __restrict__ Vt, const float* __restrict__ Vf) {
  int s = blockIdx.x, kv = blockIdx.y, d = threadIdx.x;
  Vt[(size_t)kv * HD * S_LEN + (size_t)d * S_LEN + s] =
      (__bf16)Vf[(size_t)s * (NKVH * HD) + kv * HD + d];
}

// ---- flash attention with block-causal bias --------------------------------
// One wave per (head, 16-query strip).  Scores computed TRANSPOSED
// (S^T = K_tile @ Q^T) so the exp'd probabilities convert in-lane into the
// bf16 A-operand swizzle for P@V (zero cross-lane data movement).
__global__ void attn_kernel(__bf16* __restrict__ AOb, const __bf16* __restrict__ Qb,
                            const __bf16* __restrict__ Kb, const __bf16* __restrict__ Vt,
                            const int* __restrict__ num_masks_p) {
  const int lane = threadIdx.x;
  const int q0 = blockIdx.x * 16;
  const int h  = blockIdx.y;
  const int kv = h >> 2;                 // GQA group = NH/NKVH = 4
  const int bs = *num_masks_p + 1;       // block size of the bias pattern

  const __bf16* Qh = Qb + (size_t)h  * S_LEN * HD;
  const __bf16* Kh = Kb + (size_t)kv * S_LEN * HD;
  const __bf16* Vh = Vt + (size_t)kv * HD * S_LEN;

  // Q^T B-operands for both 32-wide d-chunks (loaded once)
  const v16bf qb0 = load_frag_b(Qh, q0, HD, 0,  lane);
  const v16bf qb1 = load_frag_b(Qh, q0, HD, 32, lane);

  v8f   o[4] = {{}, {}, {}, {}};
  float m = -1e30f, l = 0.f;
  const int q   = q0 + (lane & 15);      // this lane's query (score-C column)
  const int kend = q0 + 16;              // causal bound (mask trims the rest)

  for (int kb = 0; kb < kend; kb += 32) {
    // --- transposed score tiles: C0 = keys kb..kb+15, C1 = kb+16..kb+31 ---
    v8f c0 = {}, c1 = {};
    v16bf a;
    a = load_frag_a(Kh, kb,      HD, 0,  lane); c0 = WMMA_BF16(a, qb0, c0);
    a = load_frag_a(Kh, kb,      HD, 32, lane); c0 = WMMA_BF16(a, qb1, c0);
    a = load_frag_a(Kh, kb + 16, HD, 0,  lane); c1 = WMMA_BF16(a, qb0, c1);
    a = load_frag_a(Kh, kb + 16, HD, 32, lane); c1 = WMMA_BF16(a, qb1, c1);

    // --- block-causal bias + tile max (keys = rows of C layout) ---
    const int krow = kb + ((lane >> 4) << 3);
    float p0[8], p1[8], mt = -1e30f;
#pragma unroll
    for (int r = 0; r < 8; ++r) {
      int k1 = krow + r;
      bool ok1 = (k1 <= q) && (((k1 % bs) == 0) || ((k1 / bs) == (q / bs)));
      p0[r] = ok1 ? c0[r] : -1e30f;
      int k2 = k1 + 16;
      bool ok2 = (k2 <= q) && (((k2 % bs) == 0) || ((k2 / bs) == (q / bs)));
      p1[r] = ok2 ? c1[r] : -1e30f;
      mt = fmaxf(mt, fmaxf(p0[r], p1[r]));
    }
    mt = fmaxf(mt, __shfl_xor(mt, 16));  // lanes L and L^16 share the same query

    // --- online softmax update (per query) ---
    const float mn = fmaxf(m, mt);
    const float alpha = __expf(m - mn);
    float ls = 0.f;
    v16bf pf;                            // P in A-operand swizzle, in-lane
#pragma unroll
    for (int r = 0; r < 8; ++r) {
      float e0 = __expf(p0[r] - mn);
      float e1 = __expf(p1[r] - mn);
      ls += e0 + e1;
      pf[r]     = (__bf16)e0;            // keys kb + r + 8*(lane/16)
      pf[r + 8] = (__bf16)e1;            // keys kb+16 + r + 8*(lane/16)
    }
    ls += __shfl_xor(ls, 16);
    l = l * alpha + ls;
    m = mn;

    // --- rescale accumulator: O rows are queries r + 8*(lane/16) ---
    const int rq = (lane >> 4) << 3;
    float ar[8];
#pragma unroll
    for (int r = 0; r < 8; ++r) ar[r] = __shfl(alpha, rq + r);
#pragma unroll
    for (int t = 0; t < 4; ++t)
#pragma unroll
      for (int r = 0; r < 8; ++r) o[t][r] *= ar[r];

    // --- O += P @ V  (B-operand from d-major Vt, contiguous in key) ---
#pragma unroll
    for (int t = 0; t < 4; ++t) {
      v16bf vb = load_frag_b(Vh, t * 16, S_LEN, kb, lane);
      o[t] = WMMA_BF16(pf, vb, o[t]);
    }
  }

  // --- normalize and store bf16 rows into [S][NH*HD] for the Wo GEMM ---
  const int rq = (lane >> 4) << 3;
  float lr[8];
#pragma unroll
  for (int r = 0; r < 8; ++r) lr[r] = __shfl(l, rq + r);
  const int cn = lane & 15;
#pragma unroll
  for (int t = 0; t < 4; ++t)
#pragma unroll
    for (int r = 0; r < 8; ++r)
      AOb[(size_t)(q0 + rq + r) * (NH * HD) + h * HD + t * 16 + cn] =
          (__bf16)(o[t][r] / lr[r]);
}

// ---------------------------------------------------------------------------
extern "C" void kernel_launch(void* const* d_in, const int* in_sizes, int n_in,
                              void* d_out, int out_size, void* d_ws, size_t ws_size,
                              hipStream_t stream) {
  const float* X    = (const float*)d_in[0];   // (1,S,HID)
  const float* cosp = (const float*)d_in[1];   // (1,S,D)
  const float* sinp = (const float*)d_in[2];   // (1,S,D)
  const float* Wq   = (const float*)d_in[3];   // (HID, H*D)
  const float* Wk   = (const float*)d_in[4];   // (HID, KVH*D)
  const float* Wv   = (const float*)d_in[5];   // (HID, KVH*D)
  const float* Wo   = (const float*)d_in[6];   // (H*D, HID)
  const int*   nm   = (const int*)d_in[7];     // num_masks
  float* out = (float*)d_out;                  // (1,S,HID)

  // workspace carve-up (256B aligned)
  char* ws = (char*)d_ws;
  size_t off = 0;
  auto carve = [&](size_t bytes) {
    char* p = ws + off;
    off = (off + bytes + 255) & ~(size_t)255;
    return p;
  };
  __bf16* Xb  = (__bf16*)carve((size_t)S_LEN * HIDN * 2);          // X bf16
  __bf16* Wqt = (__bf16*)carve((size_t)HIDN * (NH * HD) * 2);      // Wq^T
  __bf16* Wkt = (__bf16*)carve((size_t)HIDN * (NKVH * HD) * 2);    // Wk^T
  __bf16* Wvt = (__bf16*)carve((size_t)HIDN * (NKVH * HD) * 2);    // Wv^T
  __bf16* Wot = (__bf16*)carve((size_t)(NH * HD) * HIDN * 2);      // Wo^T
  float*  Qf  = (float*)carve((size_t)S_LEN * (NH * HD) * 4);
  float*  Kf  = (float*)carve((size_t)S_LEN * (NKVH * HD) * 4);
  float*  Vf  = (float*)carve((size_t)S_LEN * (NKVH * HD) * 4);
  __bf16* Qb  = (__bf16*)carve((size_t)NH * S_LEN * HD * 2);
  __bf16* Kb  = (__bf16*)carve((size_t)NKVH * S_LEN * HD * 2);
  __bf16* Vt  = (__bf16*)carve((size_t)NKVH * HD * S_LEN * 2);
  __bf16* AOb = (__bf16*)carve((size_t)S_LEN * (NH * HD) * 2);
  (void)ws_size; (void)n_in; (void)in_sizes; (void)out_size;

  // 1) convert inputs to bf16 (weights transposed to [N][K], coalesced)
  {
    int n = S_LEN * HIDN;
    cvt_bf16_kernel<<<(n + 255) / 256, 256, 0, stream>>>(Xb, X, n);
  }
  {
    dim3 blk(32, 8);
    tcvt_bf16_kernel<<<dim3((NH * HD) / 32, HIDN / 32), blk, 0, stream>>>(Wqt, Wq, NH * HD, HIDN);
    tcvt_bf16_kernel<<<dim3((NKVH * HD) / 32, HIDN / 32), blk, 0, stream>>>(Wkt, Wk, NKVH * HD, HIDN);
    tcvt_bf16_kernel<<<dim3((NKVH * HD) / 32, HIDN / 32), blk, 0, stream>>>(Wvt, Wv, NKVH * HD, HIDN);
    tcvt_bf16_kernel<<<dim3(HIDN / 32, (NH * HD) / 32), blk, 0, stream>>>(Wot, Wo, HIDN, NH * HD);
  }

  // 2) QKV projections (WMMA bf16, f32 accumulate, LDS-staged B)
  {
    dim3 blk(32, 8);
    gemm_bt_kernel<<<dim3((NH * HD) / 64, S_LEN / 128), blk, 0, stream>>>(
        Qf, Xb, Wqt, S_LEN, NH * HD, HIDN);
    gemm_bt_kernel<<<dim3((NKVH * HD) / 64, S_LEN / 128), blk, 0, stream>>>(
        Kf, Xb, Wkt, S_LEN, NKVH * HD, HIDN);
    gemm_bt_kernel<<<dim3((NKVH * HD) / 64, S_LEN / 128), blk, 0, stream>>>(
        Vf, Xb, Wvt, S_LEN, NKVH * HD, HIDN);
  }

  // 3) RoPE (+ D^-0.5 folded into Q) and V transpose, all repacked bf16
  rope_q_kernel<<<dim3(S_LEN, NH), HD, 0, stream>>>(Qb, Qf, cosp, sinp);
  rope_k_kernel<<<dim3(S_LEN, NKVH), HD, 0, stream>>>(Kb, Kf, cosp, sinp);
  vtrans_kernel<<<dim3(S_LEN, NKVH), HD, 0, stream>>>(Vt, Vf);

  // 4) flash attention with block-causal bias
  attn_kernel<<<dim3(S_LEN / 16, NH), 32, 0, stream>>>(AOb, Qb, Kb, Vt, nm);

  // 5) output projection
  {
    dim3 blk(32, 8);
    gemm_bt_kernel<<<dim3(HIDN / 64, S_LEN / 128), blk, 0, stream>>>(
        out, AOb, Wot, S_LEN, HIDN, NH * HD);
  }
}
